// Attention_8641474199552
// MI455X (gfx1250) — compile-verified, matrix operand order checked by reading, not run on hardware
//
#include <hip/hip_runtime.h>
#include <hip/hip_bf16.h>
#include <stdint.h>

// Shapes from the reference
#define B_      8
#define TQ      1024
#define NN      2048
#define LSTM_D  1024
#define ATT_D   128
#define EMB_D   512
#define TEMP_   0.08f
#define NB      64      // N-chunk per flash-attention iteration
#define NCH     (NN / NB)
#define KLD     136     // LDS row stride (halves) for K tiles: 256B row + 16B TDM pad
#define VLD     72      // LDS row stride for transposed V: 128B row + 16B TDM pad
#define PLD     72      // LDS row stride for P tiles
#define KCH     (64 * KLD)      // u16 per K buffer
#define VCH     (512 * VLD)     // u16 per V buffer

typedef __attribute__((ext_vector_type(16))) __bf16 v16bf;
typedef __attribute__((ext_vector_type(8)))  float  v8f;
typedef __attribute__((ext_vector_type(4)))  unsigned int v4u;
typedef __attribute__((ext_vector_type(8)))  int v8i;
typedef __attribute__((ext_vector_type(4)))  int v4i;

// fp32 -> bf16 (round-to-nearest-even) as raw u16.
__device__ __forceinline__ unsigned short f2bf(float x) {
    union { float f; unsigned u; } c; c.f = x;
    unsigned r = c.u + 0x7FFFu + ((c.u >> 16) & 1u);
    return (unsigned short)(r >> 16);
}

// Low 32 bits of a flat shared pointer == wave-relative LDS byte offset.
__device__ __forceinline__ unsigned lds_off(const void* p) {
    return (unsigned)(uintptr_t)p;
}

// TDM 2D tile load: global (row-major, bf16) -> LDS with per-row padding.
// Descriptor per CDNA5 ISA §8 (D# groups 0/1; remaining groups zero for 2D).
__device__ __forceinline__ void tdm_load_2d(unsigned lds_addr, const void* gaddr,
                                            unsigned tensor_d0, unsigned tensor_d1,
                                            unsigned long long stride0_elems,
                                            unsigned tile_d0, unsigned tile_d1,
                                            unsigned pad_interval, unsigned pad_amount) {
    unsigned long long ga = (unsigned long long)(uintptr_t)gaddr;
    v4u g0;
    g0.x = 1u;                                               // count=1 (valid), user mode
    g0.y = lds_addr;                                         // lds_addr [63:32]
    g0.z = (unsigned)(ga & 0xFFFFFFFFu);                     // global_addr [95:64]
    g0.w = (unsigned)((ga >> 32) & 0x01FFFFFFu) | (2u << 30); // addr[56:32] | type=2
    v8i g1;
    g1[0] = (int)((1u << 16)                                 // data_size = 2 bytes
                | (1u << 20)                                 // pad_enable
                | (pad_interval << 22)
                | (pad_amount << 25));
    g1[1] = (int)((tensor_d0 & 0xFFFFu) << 16);              // tensor_dim0[15:0] @ bits 63:48
    g1[2] = (int)(((tensor_d0 >> 16) & 0xFFFFu) | ((tensor_d1 & 0xFFFFu) << 16));
    g1[3] = (int)(((tensor_d1 >> 16) & 0xFFFFu) | ((tile_d0 & 0xFFFFu) << 16));
    g1[4] = (int)(tile_d1 & 0xFFFFu);                        // tile_dim1; tile_dim2 = 0
    g1[5] = (int)(stride0_elems & 0xFFFFFFFFu);              // tensor_dim0_stride[31:0]
    g1[6] = (int)((stride0_elems >> 32) & 0xFFFFu);          // stride0[47:32]; stride1 = 0
    g1[7] = 0;
    v4i z4 = {0, 0, 0, 0};
    v8i z8 = {0, 0, 0, 0, 0, 0, 0, 0};
    __builtin_amdgcn_tensor_load_to_lds(g0, g1, z4, z4, z8, 0);
}

// Load one 16x32 bf16 A/B fragment from row-major u16 storage (LDS or global).
__device__ __forceinline__ v16bf load_frag_u16(const unsigned short* base,
                                               int r0, int k0, int ld, int lane) {
    const int lh = lane >> 4;
    const unsigned short* p = base + (size_t)(r0 + (lane & 15)) * ld + k0 + lh * 8;
    union { uint4 u[2]; v16bf v; } cv;
    cv.u[0] = *(const uint4*)(p);
    cv.u[1] = *(const uint4*)(p + 16);
    return cv.v;
}

__device__ __forceinline__ v8f wmma_bf16(v16bf a, v16bf b, v8f c) {
    return __builtin_amdgcn_wmma_f32_16x16x32_bf16(false, a, false, b,
                                                   (short)0, c, false, false);
}

// ---------------------------------------------------------------------------
// Kernel 0a: K fp32 -> bf16 copy (row-major unchanged).
// ---------------------------------------------------------------------------
__global__ __launch_bounds__(256)
void cvt_k_kernel(const float* __restrict__ src, unsigned short* __restrict__ dst) {
    size_t i = ((size_t)blockIdx.x * 256 + threadIdx.x) * 4;
    float4 v = *(const float4*)(src + i);
    union { unsigned short s[4]; uint2 u; } o;
    o.s[0] = f2bf(v.x); o.s[1] = f2bf(v.y); o.s[2] = f2bf(v.z); o.s[3] = f2bf(v.w);
    *(uint2*)(dst + i) = o.u;
}

// ---------------------------------------------------------------------------
// Kernel 0b: V [b][n][e] fp32 -> Vt [b][e][n] bf16 via 64x64 LDS tile transpose.
// ---------------------------------------------------------------------------
__global__ __launch_bounds__(256)
void vtrans_kernel(const float* __restrict__ V, unsigned short* __restrict__ Vt) {
    __shared__ unsigned short tile[64 * 72];
    const int bid = blockIdx.x;
    const int b  = bid >> 8;
    const int nt = (bid >> 3) & 31;
    const int et = bid & 7;
    const int n0 = nt * 64, e0 = et * 64;
    const int tid = threadIdx.x;
    #pragma unroll
    for (int i = 0; i < 16; ++i) {
        int idx = tid + i * 256;
        int r = idx >> 6, c = idx & 63;
        tile[r * 72 + c] = f2bf(V[((size_t)b * NN + n0 + r) * EMB_D + e0 + c]);
    }
    __syncthreads();
    #pragma unroll
    for (int i = 0; i < 16; ++i) {
        int idx = tid + i * 256;
        int r = idx >> 6, c = idx & 63;
        Vt[((size_t)b * EMB_D + e0 + r) * NN + n0 + c] = tile[c * 72 + r];
    }
}

// ---------------------------------------------------------------------------
// Kernel 1: Q = query @ Wq^T -> bf16 workspace. M=8192, N=128, K=1024.
// ---------------------------------------------------------------------------
__global__ __launch_bounds__(256)
void qproj_kernel(const float* __restrict__ query, const float* __restrict__ Wq,
                  unsigned short* __restrict__ Qbf) {
    __shared__ unsigned short Alds[64 * 72];
    __shared__ unsigned short Blds[128 * 72];
    const int tid  = threadIdx.x;
    const int lane = tid & 31;
    const int w    = tid >> 5;
    const int s    = w >> 1, h = w & 1;
    const int lh   = lane >> 4;
    const int row0 = blockIdx.x * 64;

    v8f acc[4] = {};

    for (int kc = 0; kc < LSTM_D / 64; ++kc) {
        #pragma unroll
        for (int i = 0; i < 16; ++i) {
            int idx = tid + i * 256;
            int r = idx >> 6, c = idx & 63;
            Alds[r * 72 + c] = f2bf(query[(size_t)(row0 + r) * LSTM_D + kc * 64 + c]);
        }
        #pragma unroll
        for (int i = 0; i < 32; ++i) {
            int idx = tid + i * 256;
            int r = idx >> 6, c = idx & 63;
            Blds[r * 72 + c] = f2bf(Wq[(size_t)r * LSTM_D + kc * 64 + c]);
        }
        __syncthreads();

        #pragma unroll
        for (int ks = 0; ks < 2; ++ks) {
            v16bf af = load_frag_u16(Alds, 16 * s, 32 * ks, 72, lane);
            #pragma unroll
            for (int t = 0; t < 4; ++t) {
                v16bf bf = load_frag_u16(Blds, 64 * h + 16 * t, 32 * ks, 72, lane);
                acc[t] = wmma_bf16(af, bf, acc[t]);
            }
        }
        __syncthreads();
    }

    #pragma unroll
    for (int t = 0; t < 4; ++t)
        #pragma unroll
        for (int j = 0; j < 8; ++j) {
            int r = row0 + 16 * s + j + 8 * lh;
            int c = 64 * h + 16 * t + (lane & 15);
            Qbf[(size_t)r * ATT_D + c] = f2bf(acc[t][j]);
        }
}

// ---------------------------------------------------------------------------
// Kernel 2: fused flash attention; K/V chunks streamed by the TDM into
// double-buffered LDS (wave 0 issues, TENSORcnt-gated, one chunk ahead).
// ---------------------------------------------------------------------------
__global__ __launch_bounds__(256)
void attn_kernel(const unsigned short* __restrict__ Qbf,
                 const unsigned short* __restrict__ Kbf,
                 const unsigned short* __restrict__ Vtw,
                 const unsigned char* __restrict__ mask,
                 const float* __restrict__ stepp, float* __restrict__ out) {
    extern __shared__ unsigned short smem[];
    // Layout (u16 units): K buf0 @0, K buf1 @KCH, V buf0 @2*KCH, V buf1 @2*KCH+VCH,
    // P @2*KCH+2*VCH. All pointers derived with runtime arithmetic (no static
    // initializers holding LDS addresses).
    unsigned short* Plds = smem + 2 * KCH + 2 * VCH;

    const int tid  = threadIdx.x;
    const int lane = tid & 31;
    const int w    = tid >> 5;
    const int s    = w >> 1, h = w & 1;
    const int lh   = lane >> 4;
    const int b    = blockIdx.x >> 4;
    const int row0 = (blockIdx.x & 15) * 64;
    const float step    = stepp[0];
    const float invnorm = 0.08838834764831845f;     // 1/sqrt(ATT_D)

    const unsigned smem0 = lds_off(smem);           // LDS byte offset of smem base

    const unsigned short* Kb  = Kbf + (size_t)b * NN * ATT_D;
    const unsigned short* Vtb = Vtw + (size_t)b * EMB_D * NN;

    // Q strip fragments live in registers for the whole kernel
    v16bf qf[4];
    {
        const unsigned short* qbase = Qbf + (size_t)(b * TQ + row0 + 16 * s) * ATT_D;
        #pragma unroll
        for (int ks = 0; ks < 4; ++ks)
            qf[ks] = load_frag_u16(qbase, 0, 32 * ks, ATT_D, lane);
    }

    // Kick off DMA for chunk 0 (wave 0 only; TDM is one op per wave).
    if (w == 0) {
        // K chunk: 64 rows x 128 elems (256B) -> 272B LDS rows (pad 4 DW every 64 DW)
        tdm_load_2d(smem0, Kb, ATT_D, NB, ATT_D, ATT_D, NB, 5, 3);
        // V chunk: 512 rows x 64 elems (128B) -> 144B LDS rows (pad 4 DW every 32 DW)
        tdm_load_2d(smem0 + 4 * KCH, Vtb, NB, EMB_D, NN, NB, EMB_D, 4, 3);
    }

    v8f acc[16] = {};
    float mrow[8], lrow[8];
    #pragma unroll
    for (int j = 0; j < 8; ++j) { mrow[j] = -INFINITY; lrow[j] = 0.f; }

    for (int jc = 0; jc < NCH; ++jc) {
        const int n0  = jc * NB;
        const int cur = jc & 1;
        const int nxt = cur ^ 1;
        const unsigned short* Klds = smem + cur * KCH;
        const unsigned short* Vt   = smem + 2 * KCH + cur * VCH;

        if (w == 0) {
            if (jc + 1 < NCH) {
                // issue next chunk into the other buffer, then wait for current
                tdm_load_2d(smem0 + (unsigned)nxt * (2 * KCH),
                            Kb + (size_t)(n0 + NB) * ATT_D,
                            ATT_D, NB, ATT_D, ATT_D, NB, 5, 3);
                tdm_load_2d(smem0 + 4 * KCH + (unsigned)nxt * (2 * VCH),
                            Vtb + n0 + NB,
                            NB, EMB_D, NN, NB, EMB_D, 4, 3);
                __builtin_amdgcn_s_wait_tensorcnt(2);   // current pair landed
            } else {
                __builtin_amdgcn_s_wait_tensorcnt(0);
            }
        }
        __syncthreads();

        // --- S = Q K^T for this wave's 16x64 strip (duplicated by wave pair) ---
        v8f S[4] = {};
        #pragma unroll
        for (int c = 0; c < 4; ++c)
            #pragma unroll
            for (int ks = 0; ks < 4; ++ks) {
                v16bf kf = load_frag_u16(Klds, 16 * c, 32 * ks, KLD, lane);
                S[c] = wmma_bf16(qf[ks], kf, S[c]);
            }

        // --- scale by 1/sqrt(d), gaussian guide, bool mask -> -inf ---
        #pragma unroll
        for (int c = 0; c < 4; ++c) {
            int n = n0 + 16 * c + (lane & 15);
            float pos = (float)(n + 1) * (1.0f / NN);
            float d = step - pos;
            float g = __expf(-d * d * (1.0f / TEMP_)) * invnorm;
            #pragma unroll
            for (int j = 0; j < 8; ++j) {
                int r = row0 + 16 * s + j + 8 * lh;
                float v = S[c][j] * g;
                if (mask[((size_t)b * TQ + r) * NN + n]) v = -INFINITY;
                S[c][j] = v;
            }
        }

        // --- online softmax (wave-local stats, 16-lane shfl reductions) ---
        float alpha[8];
        #pragma unroll
        for (int j = 0; j < 8; ++j) {
            float mx = fmaxf(fmaxf(S[0][j], S[1][j]), fmaxf(S[2][j], S[3][j]));
            #pragma unroll
            for (int off = 1; off < 16; off <<= 1)
                mx = fmaxf(mx, __shfl_xor(mx, off, 32));
            float mnew = fmaxf(mrow[j], mx);
            alpha[j] = (mrow[j] == -INFINITY) ? 0.f : __expf(mrow[j] - mnew);
            float shift = (mnew == -INFINITY) ? 0.f : mnew;
            mrow[j] = mnew;
            float ssum = 0.f;
            #pragma unroll
            for (int c = 0; c < 4; ++c) {
                float p = __expf(S[c][j] - shift);
                S[c][j] = p;
                ssum += p;
            }
            #pragma unroll
            for (int off = 1; off < 16; off <<= 1)
                ssum += __shfl_xor(ssum, off, 32);
            lrow[j] = lrow[j] * alpha[j] + ssum;
        }

        #pragma unroll
        for (int t = 0; t < 16; ++t)
            #pragma unroll
            for (int j = 0; j < 8; ++j)
                acc[t][j] *= alpha[j];

        // P (C/D layout) -> LDS row-major; one wave of each duplicate pair writes
        if (h == 0) {
            unsigned short* pb = Plds + s * 16 * PLD;
            #pragma unroll
            for (int c = 0; c < 4; ++c)
                #pragma unroll
                for (int j = 0; j < 8; ++j)
                    pb[(j + 8 * lh) * PLD + 16 * c + (lane & 15)] = f2bf(S[c][j]);
        }
        __syncthreads();

        // --- O += P V for this wave's 256-column half ---
        const unsigned short* pb = Plds + s * 16 * PLD;
        #pragma unroll
        for (int ks = 0; ks < 2; ++ks) {
            v16bf pf = load_frag_u16(pb, 0, 32 * ks, PLD, lane);
            #pragma unroll
            for (int t = 0; t < 16; ++t) {
                v16bf vf = load_frag_u16(Vt, 256 * h + 16 * t, 32 * ks, VLD, lane);
                acc[t] = wmma_bf16(pf, vf, acc[t]);
            }
        }
        __syncthreads();
    }

    // --- epilogue: out = acc / l ---
    #pragma unroll
    for (int t = 0; t < 16; ++t)
        #pragma unroll
        for (int j = 0; j < 8; ++j) {
            int r = row0 + 16 * s + j + 8 * lh;
            int e = 256 * h + 16 * t + (lane & 15);
            out[((size_t)b * TQ + r) * EMB_D + e] = acc[t][j] / lrow[j];
        }
}

// ---------------------------------------------------------------------------
// Inputs: query f32, K f32, V f32, Wq f32, step f32[1], mask bool (1B/elem).
// Workspace: [0,2MB) Q bf16; [2MB,6MB) K bf16; [6MB,22MB) Vt bf16.
// ---------------------------------------------------------------------------
extern "C" void kernel_launch(void* const* d_in, const int* in_sizes, int n_in,
                              void* d_out, int out_size, void* d_ws, size_t ws_size,
                              hipStream_t stream) {
    const float*         query = (const float*)d_in[0];
    const float*         K     = (const float*)d_in[1];
    const float*         V     = (const float*)d_in[2];
    const float*         Wq    = (const float*)d_in[3];
    const float*         step  = (const float*)d_in[4];
    const unsigned char* mask  = (const unsigned char*)d_in[5];
    float*               out   = (float*)d_out;

    unsigned short* Qbf = (unsigned short*)d_ws;                         // 8192*128
    unsigned short* Kbf = Qbf + (size_t)B_ * TQ * ATT_D;                 // 8*2048*128
    unsigned short* Vtw = Kbf + (size_t)B_ * NN * ATT_D;                 // 8*512*2048

    cvt_k_kernel<<<dim3((B_ * NN * ATT_D) / (4 * 256)), dim3(256), 0, stream>>>(K, Kbf);
    vtrans_kernel<<<dim3(B_ * 32 * 8), dim3(256), 0, stream>>>(V, Vtw);
    qproj_kernel<<<dim3((B_ * TQ) / 64), dim3(256), 0, stream>>>(query, Wq, Qbf);

    size_t shmem = (size_t)(2 * KCH + 2 * VCH + 4 * 16 * PLD) * sizeof(unsigned short);
    attn_kernel<<<dim3(B_ * (TQ / 64)), dim3(256), shmem, stream>>>(Qbf, Kbf, Vtw, mask, step, out);
}
